// MultiHeadedAttentionBlock_75952201662611
// MI455X (gfx1250) — matrix operand order repaired and checked by reading, not compiled, not run on hardware
//
#include <hip/hip_runtime.h>
#include <cstdint>
#include <cstddef>

// ---------------------------------------------------------------------------
// Problem constants (fixed by the reference)
// ---------------------------------------------------------------------------
#define B_   4
#define S_   2048
#define D_   1024
#define H_   16
#define HD_  64
#define M_   (B_ * S_)        // 8192 token rows

typedef __bf16 bf16_t;
typedef __attribute__((ext_vector_type(16))) __bf16 v16bf;
typedef __attribute__((ext_vector_type(8)))  float  v8f;
typedef unsigned int u32x4 __attribute__((ext_vector_type(4)));
typedef int          i32x4 __attribute__((ext_vector_type(4)));
typedef int          i32x8 __attribute__((ext_vector_type(8)));

#if __has_builtin(__builtin_amdgcn_tensor_load_to_lds) && \
    __has_builtin(__builtin_amdgcn_s_wait_tensorcnt)
#define USE_TDM 1
#else
#define USE_TDM 0
#endif

union Frag32B { v16bf v; uint4 u[2]; };

// ---------------------------------------------------------------------------
// WMMA fragment loaders (CDNA5 ISA 7.12.2, wave32)
// A (16x32 bf16): lane L holds row M = L%16; two 16B chunks at K = k0+g*8 and
//                 K = k0+16+g*8 (g = L/16).
// B (32x16 bf16): lane L holds column N = L%16; two 16B chunks at K = k0+g*8
//                 and K = k0+16+g*8. "base" is the matrix whose ROWS are B's columns.
// ---------------------------------------------------------------------------
static __device__ __forceinline__
v16bf load_a_frag(const bf16_t* base, int ld, int row0, int k0, int lane) {
    const int r = row0 + (lane & 15);
    const int g = lane >> 4;
    const bf16_t* p = base + (size_t)r * ld + (k0 + g * 8);
    Frag32B f;
    f.u[0] = *(const uint4*)(p);
    f.u[1] = *(const uint4*)(p + 16);
    return f.v;
}

static __device__ __forceinline__
v16bf load_b_frag(const bf16_t* base, int ld, int col0, int k0, int lane) {
    const int n = col0 + (lane & 15);
    const int g = lane >> 4;
    const bf16_t* p = base + (size_t)n * ld + (k0 + g * 8);
    Frag32B f;
    f.u[0] = *(const uint4*)(p);
    f.u[1] = *(const uint4*)(p + 16);
    return f.v;
}

static __device__ __forceinline__
v8f wmma_bf16(v16bf a, v16bf b, v8f c) {
    return __builtin_amdgcn_wmma_f32_16x16x32_bf16(
        false, a, false, b, (short)0, c, false, false);
}

// ---------------------------------------------------------------------------
// fp32 -> bf16 conversion, 8 elements / lane
// ---------------------------------------------------------------------------
__global__ __launch_bounds__(256)
void cvt_f32_bf16(const float* __restrict__ s, bf16_t* __restrict__ d, long n) {
    long i = ((long)blockIdx.x * blockDim.x + threadIdx.x) * 8;
    if (i + 8 > n) return;
    float4 a = *(const float4*)(s + i);
    float4 b = *(const float4*)(s + i + 4);
    union { bf16_t h[8]; uint4 u; } o;
    o.h[0] = (bf16_t)a.x; o.h[1] = (bf16_t)a.y; o.h[2] = (bf16_t)a.z; o.h[3] = (bf16_t)a.w;
    o.h[4] = (bf16_t)b.x; o.h[5] = (bf16_t)b.y; o.h[6] = (bf16_t)b.z; o.h[7] = (bf16_t)b.w;
    *(uint4*)(d + i) = o.u;
}

// ---------------------------------------------------------------------------
// GEMM: Y = A(MxK) @ W^T + bias, W stored (N,K) row-major. K = N = D_ = 1024.
//
// 4 waves / workgroup, tile 128x64. The 64-row W slab is staged through LDS
// in double-buffered 64x128-element chunks (16KB each) filled by the Tensor
// Data Mover (TENSOR_LOAD_TO_LDS, 2D D#), synchronized with s_wait_tensorcnt
// + workgroup barriers. Each wave computes 32x64: per 32-k-step it loads all
// four B fragments (one ds clause), two A fragments (global clause), then
// issues 8 back-to-back WMMAs -- keeps the XDL pipe streaming (ISA 7.5).
//
// mode 0: bf16 out, (B,H,S,HD)    [Q, K projections]
// mode 1: bf16 out, (B,H,HD,S)    [V projection -> V^T]
// mode 2: fp32 out, (M,N)+bias    [final projection -> d_out]
// ---------------------------------------------------------------------------
#define KC_ 128                       // k-chunk staged per buffer
__global__ __launch_bounds__(128)
void gemm_bf16(const bf16_t* __restrict__ A, const bf16_t* __restrict__ W,
               const float* __restrict__ bias,
               bf16_t* __restrict__ outb, float* __restrict__ outf, int mode) {
    __shared__ __align__(16) bf16_t Wt[2][64 * KC_];   // 32 KB double buffer

    const int K    = D_;
    const int tid  = threadIdx.x;
    const int lane = tid & 31;
    const int wave = tid >> 5;
    const int m0   = blockIdx.x * 128 + wave * 32;
    const int n0   = blockIdx.y * 64;

    v8f acc[2][4] = {};

#if USE_TDM
    // ---- Tensor DMA descriptor (ISA 8.3/8.4), 2D: W slab rows n0..n0+63 ----
    i32x8 g1;
    g1[0] = 0x00010000;                                   // data_size = 1 -> 2B
    g1[1] = (int)((K & 0xFFFF) << 16);                    // tensor_dim0[15:0]
    g1[2] = (int)((unsigned)(K >> 16) | ((D_ & 0xFFFF) << 16)); // dim0 hi | dim1 lo
    g1[3] = (int)((unsigned)(D_ >> 16) | ((unsigned)KC_ << 16)); // dim1 hi | tile_dim0
    g1[4] = 64;                                           // tile_dim1
    g1[5] = K;                                            // tensor_dim0_stride
    g1[6] = 0;
    g1[7] = 0;
    const i32x4 gz4 = {0, 0, 0, 0};                       // groups 2/3: NULL (2D)
    const i32x8 gz8 = {0, 0, 0, 0, 0, 0, 0, 0};           // toolchain's extra arg
    const uint64_t wbase = (uint64_t)(uintptr_t)(W + (size_t)n0 * K);
    const uint32_t lds0  = (uint32_t)(uintptr_t)(&Wt[0][0]);  // LDS offset (low 32b)

    auto tdm_issue = [&](int kc, int buf) {
        const uint64_t ga = wbase + (uint64_t)kc * 2;     // tile start address
        u32x4 g0;
        g0[0] = 1u;                                       // count = 1, user mode
        g0[1] = lds0 + (uint32_t)buf * (64 * KC_ * 2);    // lds_addr
        g0[2] = (uint32_t)ga;                             // global_addr[31:0]
        g0[3] = ((uint32_t)(ga >> 32) & 0x01FFFFFFu)      // global_addr[56:32]
                | 0x80000000u;                            // type = 2 ("image")
        __builtin_amdgcn_tensor_load_to_lds(g0, g1, gz4, gz4, gz8, 0);
    };

    if (wave == 0) tdm_issue(0, 0);
#endif

    for (int kc = 0; kc < K; kc += KC_) {
        const int cur = (kc >> 7) & 1;
#if USE_TDM
        if (wave == 0) {
            if (kc + KC_ < K) {
                tdm_issue(kc + KC_, cur ^ 1);             // prefetch next chunk
                __builtin_amdgcn_s_wait_tensorcnt(1);     // current chunk done
            } else {
                __builtin_amdgcn_s_wait_tensorcnt(0);
            }
        }
        __syncthreads();
#else
        {   // cooperative fill: 64 rows x 128 halves = 1024 uint4 / 128 threads
            const int row = tid >> 1;
            const uint4* src = (const uint4*)(W + (size_t)(n0 + row) * K + kc)
                               + (tid & 1) * 8;
            uint4* dst = (uint4*)(&Wt[cur][row * KC_]) + (tid & 1) * 8;
#pragma unroll
            for (int i = 0; i < 8; ++i) dst[i] = src[i];
        }
        __syncthreads();
#endif

#pragma unroll
        for (int ks = 0; ks < KC_ / 32; ++ks) {
            const int k0 = kc + ks * 32;
            // batch all loads first: one global clause + one ds clause,
            // then 8 WMMAs back-to-back
            v16bf a0 = load_a_frag(A, K, m0,      k0, lane);
            v16bf a1 = load_a_frag(A, K, m0 + 16, k0, lane);
            v16bf b[4];
#pragma unroll
            for (int nt = 0; nt < 4; ++nt)
                b[nt] = load_b_frag(&Wt[cur][0], KC_, nt * 16, ks * 32, lane);
#pragma unroll
            for (int nt = 0; nt < 4; ++nt) {
                acc[0][nt] = wmma_bf16(a0, b[nt], acc[0][nt]);
                acc[1][nt] = wmma_bf16(a1, b[nt], acc[1][nt]);
            }
        }
        __syncthreads();   // all waves done reading before buffer is re-filled
    }

    const int cl = lane & 15;
    const int g  = lane >> 4;
#pragma unroll
    for (int mt = 0; mt < 2; ++mt) {
#pragma unroll
        for (int nt = 0; nt < 4; ++nt) {
            const int n  = n0 + nt * 16 + cl;
            const float bn = bias[n];
#pragma unroll
            for (int r = 0; r < 8; ++r) {
                const int m = m0 + mt * 16 + r + g * 8;
                const float val = acc[mt][nt][r] + bn;
                if (mode == 2) {
                    outf[(size_t)m * D_ + n] = val;
                } else {
                    const int b_  = m / S_,  s  = m % S_;
                    const int h   = n / HD_, hd = n % HD_;
                    const size_t idx = (mode == 0)
                        ? (((size_t)(b_ * H_ + h) * S_  + s ) * HD_ + hd)
                        : (((size_t)(b_ * H_ + h) * HD_ + hd) * S_  + s );
                    outb[idx] = (bf16_t)val;
                }
            }
        }
    }
}

// ---------------------------------------------------------------------------
// Flash attention, one wave per (b*H+h, 16-query-row tile).
// Per 32-key step: 4 score WMMAs (batched K-fragment loads), fp32 online
// softmax (shfl-xor width-16 row reductions, exp2 on the TRANS pipe), P
// staged bf16 through LDS, 4 PV WMMAs against V^T (batched V-fragment loads).
// Prefetch hints pull the next K/V tiles toward the caches.
// ---------------------------------------------------------------------------
__global__ __launch_bounds__(32)
void attn_flash(const bf16_t* __restrict__ Qh, const bf16_t* __restrict__ Kh,
                const bf16_t* __restrict__ VT, const int* __restrict__ mask,
                bf16_t* __restrict__ O) {
    __shared__ __align__(16) bf16_t Pt[16 * 48];  // row stride 96B: aligned b128

    const int lane = threadIdx.x;
    const int qt   = blockIdx.x;
    const int bh   = blockIdx.y;
    const int b    = bh / H_;
    const int h    = bh % H_;
    const int cl   = lane & 15;
    const int g    = lane >> 4;

    const bf16_t* Qb = Qh + (size_t)bh * S_ * HD_;
    const bf16_t* Kb = Kh + (size_t)bh * S_ * HD_;
    const bf16_t* Vb = VT + (size_t)bh * HD_ * S_;
    const int*    qm = mask + (size_t)b * S_;

    const v16bf aq0 = load_a_frag(Qb, HD_, qt * 16, 0,  lane);
    const v16bf aq1 = load_a_frag(Qb, HD_, qt * 16, 32, lane);

    v8f o[4] = {};
    float mrow[8], lrow[8];
#pragma unroll
    for (int r = 0; r < 8; ++r) { mrow[r] = -__builtin_inff(); lrow[r] = 0.f; }

    const float C = 0.125f * 1.44269504088896340736f;  // (1/sqrt(HD)) * log2(e)

    for (int kb = 0; kb < S_; kb += 32) {
        if (kb + 32 < S_) {   // global_prefetch_b8 for the next key/value tiles
            __builtin_prefetch(Kb + (size_t)(kb + 32 + cl) * HD_, 0, 0);
            __builtin_prefetch(Vb + (size_t)cl * S_ + kb + 32, 0, 0);
        }

        // batch the 4 K-fragment loads (one clause), then 4 WMMAs
        v16bf bk00 = load_b_frag(Kb, HD_, kb,      0,  lane);
        v16bf bk01 = load_b_frag(Kb, HD_, kb,      32, lane);
        v16bf bk10 = load_b_frag(Kb, HD_, kb + 16, 0,  lane);
        v16bf bk11 = load_b_frag(Kb, HD_, kb + 16, 32, lane);

        v8f s0 = {}, s1 = {};
        s0 = wmma_bf16(aq0, bk00, s0);
        s0 = wmma_bf16(aq1, bk01, s0);
        s1 = wmma_bf16(aq0, bk10, s1);
        s1 = wmma_bf16(aq1, bk11, s1);

        const bool km0 = qm[kb + cl]      != 0;
        const bool km1 = qm[kb + 16 + cl] != 0;

        float alpha[8];
#pragma unroll
        for (int r = 0; r < 8; ++r) {
            float v = fmaxf(km0 ? s0[r] : -__builtin_inff(),
                            km1 ? s1[r] : -__builtin_inff());
            v = fmaxf(v, __shfl_xor(v, 1, 16));
            v = fmaxf(v, __shfl_xor(v, 2, 16));
            v = fmaxf(v, __shfl_xor(v, 4, 16));
            v = fmaxf(v, __shfl_xor(v, 8, 16));
            const float mn = fmaxf(mrow[r], v);

            const float p0 = km0 ? exp2f((s0[r] - mn) * C) : 0.f;
            const float p1 = km1 ? exp2f((s1[r] - mn) * C) : 0.f;

            Pt[(r + g * 8) * 48 + cl]      = (bf16_t)p0;
            Pt[(r + g * 8) * 48 + 16 + cl] = (bf16_t)p1;

            float ps = p0 + p1;
            ps += __shfl_xor(ps, 1, 16);
            ps += __shfl_xor(ps, 2, 16);
            ps += __shfl_xor(ps, 4, 16);
            ps += __shfl_xor(ps, 8, 16);

            alpha[r] = (mrow[r] == -__builtin_inff()) ? 0.f
                                                      : exp2f((mrow[r] - mn) * C);
            lrow[r]  = lrow[r] * alpha[r] + ps;
            mrow[r]  = mn;
        }

#pragma unroll
        for (int nt = 0; nt < 4; ++nt)
#pragma unroll
            for (int r = 0; r < 8; ++r)
                o[nt][r] = o[nt][r] * alpha[r];

        // P A-fragment from LDS (same-wave LDS ops are ordered) + batched V
        const v16bf ap = load_a_frag(Pt, 48, 0, 0, lane);
        v16bf bv[4];
#pragma unroll
        for (int nt = 0; nt < 4; ++nt)
            bv[nt] = load_b_frag(Vb, S_, nt * 16, kb, lane);
#pragma unroll
        for (int nt = 0; nt < 4; ++nt)
            o[nt] = wmma_bf16(ap, bv[nt], o[nt]);
    }

#pragma unroll
    for (int r = 0; r < 8; ++r) {
        const int q = qt * 16 + r + g * 8;
        float inv = (lrow[r] > 0.f) ? (1.f / lrow[r]) : 0.f;
        if (qm[q] == 0) inv = 0.f;
#pragma unroll
        for (int nt = 0; nt < 4; ++nt) {
            const int dcol = h * HD_ + nt * 16 + cl;
            O[((size_t)b * S_ + q) * D_ + dcol] = (bf16_t)(o[nt][r] * inv);
        }
    }
}

// ---------------------------------------------------------------------------
// Host-side orchestration (graph-capture safe: only kernel launches on stream)
// ---------------------------------------------------------------------------
extern "C" void kernel_launch(void* const* d_in, const int* in_sizes, int n_in,
                              void* d_out, int out_size, void* d_ws, size_t ws_size,
                              hipStream_t stream) {
    (void)in_sizes; (void)n_in; (void)out_size; (void)ws_size;

    const float* q    = (const float*)d_in[0];
    const int*   qmsk = (const int*)  d_in[1];
    const float* Wq   = (const float*)d_in[2];
    const float* bq   = (const float*)d_in[3];
    const float* Wk   = (const float*)d_in[4];
    const float* bk   = (const float*)d_in[5];
    const float* Wv   = (const float*)d_in[6];
    const float* bv   = (const float*)d_in[7];
    const float* Wo   = (const float*)d_in[8];
    const float* bo   = (const float*)d_in[9];
    float* out = (float*)d_out;

    char*  ws  = (char*)d_ws;
    size_t off = 0;
    auto carve = [&](size_t elems) {
        void* p = ws + off;
        off += (elems * sizeof(bf16_t) + 255) & ~(size_t)255;
        return (bf16_t*)p;
    };
    bf16_t* Xb  = carve((size_t)M_ * D_);
    bf16_t* Wqb = carve((size_t)D_ * D_);
    bf16_t* Wkb = carve((size_t)D_ * D_);
    bf16_t* Wvb = carve((size_t)D_ * D_);
    bf16_t* Wob = carve((size_t)D_ * D_);
    bf16_t* Qhb = carve((size_t)M_ * D_);   // (B,H,S,HD)
    bf16_t* Khb = carve((size_t)M_ * D_);   // (B,H,S,HD)
    bf16_t* VTb = carve((size_t)M_ * D_);   // (B,H,HD,S)
    bf16_t* Ob  = carve((size_t)M_ * D_);   // (B,S,D)

    const long nX = (long)M_ * D_;
    const long nW = (long)D_ * D_;
    cvt_f32_bf16<<<(int)(nX / 2048), 256, 0, stream>>>(q,  Xb,  nX);
    cvt_f32_bf16<<<(int)(nW / 2048), 256, 0, stream>>>(Wq, Wqb, nW);
    cvt_f32_bf16<<<(int)(nW / 2048), 256, 0, stream>>>(Wk, Wkb, nW);
    cvt_f32_bf16<<<(int)(nW / 2048), 256, 0, stream>>>(Wv, Wvb, nW);
    cvt_f32_bf16<<<(int)(nW / 2048), 256, 0, stream>>>(Wo, Wob, nW);

    const dim3 gg(M_ / 128, D_ / 64);       // (64, 16) workgroups of 4 waves
    gemm_bf16<<<gg, 128, 0, stream>>>(Xb, Wqb, bq, Qhb, nullptr, 0);
    gemm_bf16<<<gg, 128, 0, stream>>>(Xb, Wkb, bk, Khb, nullptr, 0);
    gemm_bf16<<<gg, 128, 0, stream>>>(Xb, Wvb, bv, VTb, nullptr, 1);

    const dim3 ga(S_ / 16, B_ * H_);        // (128, 64) single-wave workgroups
    attn_flash<<<ga, 32, 0, stream>>>(Qhb, Khb, VTb, qmsk, Ob);

    gemm_bf16<<<gg, 128, 0, stream>>>(Ob, Wob, bo, nullptr, out, 2);
}
